// ATGCO_V1_16303695856135
// MI455X (gfx1250) — matrix operand order repaired and checked by reading, not compiled, not run on hardware
//
#include <hip/hip_runtime.h>
#include <hip/hip_bf16.h>

// ---------------------------------------------------------------------------
// Problem: out[n,e,a] = sum_{k,d,m} adj[n,m,k] * x[m,d,(a-k)%4] * weight[d,e,k]
// Rewritten as ONE GEMM:  C(4096x128) = A(4096x16384) @ B(16384x128)
//   A  = adj viewed flat (row-major, col c = m*4+k)  -- 256 MB streamed once
//   B[c = m*4+k, col = e*4+a] = sum_d x[m,d,(a-k)&3] * weight[d,e,k]  (4 MB bf16)
//   C row-major == required (NODES, DOUT, N_REL) output layout.
// bf16 WMMA (16x16x32) keeps the kernel on the HBM roof (~11 us @ 23.3 TB/s);
// fp32 WMMA (16x16x4) would be ~8x lower FLOP/instr and compute-bound.
// A is cvt'd fp32->bf16 in registers (HBM traffic stays exactly 256 MB);
// B panels are moved with GLOBAL_LOAD_ASYNC_TO_LDS_B128 (ASYNCcnt path).
// Workspace requirement: 4 MB (bf16 B matrix).
// ---------------------------------------------------------------------------

typedef __attribute__((ext_vector_type(16))) __bf16 v16bf;
typedef __attribute__((ext_vector_type(8)))  __bf16 v8bf;
typedef __attribute__((ext_vector_type(8)))  float  v8f;
typedef __attribute__((ext_vector_type(4)))  int    v4i;

#define M_DIM   4096
#define K_DIM   16384          // NODES * N_REL
#define NCOL    128            // DOUT * N_REL
#define M_BLK   64
#define KSPLIT  2
#define KC      (K_DIM / KSPLIT)   // 8192
#define NKP     (KC / 32)          // 256 k-panels per block

#if defined(__has_builtin)
#  if __has_builtin(__builtin_amdgcn_global_load_async_to_lds_b128)
#    define HAVE_ASYNC_LDS 1
#  else
#    define HAVE_ASYNC_LDS 0
#  endif
#  if __has_builtin(__builtin_amdgcn_s_wait_asynccnt)
#    define HAVE_WAIT_ASYNC 1
#  else
#    define HAVE_WAIT_ASYNC 0
#  endif
#else
#  define HAVE_ASYNC_LDS 0
#  define HAVE_WAIT_ASYNC 0
#endif

#if HAVE_ASYNC_LDS
typedef __attribute__((address_space(1))) v4i gv4i;   // global int4
typedef __attribute__((address_space(3))) v4i lv4i;   // LDS int4
#endif

// Async 16B global->LDS copy (ASYNCcnt path); sync fallback if builtin absent.
__device__ __forceinline__ void copy_b128_to_lds(const void* g, void* l) {
#if HAVE_ASYNC_LDS
    // inttoptr casts: flat == as1 numerically; flat LDS address truncates to
    // the LDS offset (ISA 10.2 aperture rules).
    __builtin_amdgcn_global_load_async_to_lds_b128(
        (gv4i*)(size_t)g, (lv4i*)(unsigned int)(size_t)l, 0, 0);
#else
    *(int4*)l = *(const int4*)g;
#endif
}

__device__ __forceinline__ void wait_async_lds() {
#if HAVE_ASYNC_LDS
#  if HAVE_WAIT_ASYNC
    __builtin_amdgcn_s_wait_asynccnt(0);
#  else
    asm volatile("s_wait_asynccnt 0" ::: "memory");
#  endif
#endif
}

// ---------------------------------------------------------------------------
// Phase 1: build B (16384 x 128) in bf16, panel-major [k>>5][col][k&31],
//          and zero d_out (phase 2 accumulates with atomics across K-chunks).
// ---------------------------------------------------------------------------
__global__ __launch_bounds__(256) void atgco_build_B(
    const float* __restrict__ x,      // [4096][32][4]
    const float* __restrict__ w,      // [32][32][4]
    __bf16* __restrict__ Bws,         // [512][128][32] bf16
    float* __restrict__ out)          // [4096][128] f32
{
    const int t   = blockIdx.x * blockDim.x + threadIdx.x;  // 0 .. 2M-1
    const int col = t & 127;          // e*4 + a
    const int k   = t >> 7;           // m*4 + krel
    const int m    = k >> 2;
    const int krel = k & 3;
    const int e    = col >> 2;
    const int a    = col & 3;
    const int j    = (a - krel) & 3;  // x relation index

    const float* xr = x + m * 128 + j;        // x[m, d, j], stride 4 over d
    const float* wr = w + e * 4 + krel;       // weight[d, e, krel], stride 128 over d
    float acc = 0.f;
#pragma unroll
    for (int d = 0; d < 32; ++d)
        acc += xr[d * 4] * wr[d * 128];

    Bws[(k >> 5) * (NCOL * 32) + col * 32 + (k & 31)] = (__bf16)acc;

    if (t < M_DIM * NCOL) out[t] = 0.f;
}

__device__ __forceinline__ v8bf pack8(float4 lo, float4 hi) {
    v8bf r;
    r[0] = (__bf16)lo.x; r[1] = (__bf16)lo.y; r[2] = (__bf16)lo.z; r[3] = (__bf16)lo.w;
    r[4] = (__bf16)hi.x; r[5] = (__bf16)hi.y; r[6] = (__bf16)hi.z; r[7] = (__bf16)hi.w;
    return r;
}

// ---------------------------------------------------------------------------
// Phase 2: C += A_chunk @ B_chunk using v_wmma_f32_16x16x32_bf16.
// grid = (M_DIM/M_BLK, KSPLIT), block = 256 threads (8 wave32s).
// Each wave computes a 2x2 grid of 16x16 output tiles (32 acc VGPRs).
// Double-buffered LDS, ONE barrier per k-panel:
//   reads of buf[nxt] finished before the barrier at the end of iter kp-1,
//   so writes into buf[nxt] are safe from the top of iter kp.
// ---------------------------------------------------------------------------
__global__ __launch_bounds__(256) void atgco_gemm_bf16(
    const float*  __restrict__ A,     // adj flat: [4096][16384] f32
    const __bf16* __restrict__ Bws,   // [512][128][32] bf16
    float* __restrict__ out)          // [4096][128] f32
{
    __shared__ __align__(16) __bf16 Ash[2][M_BLK * 32];   //  8 KB
    __shared__ __align__(16) __bf16 Bsh[2][NCOL * 32];    // 16 KB

    const int tid     = threadIdx.x;
    const int m_base  = blockIdx.x * M_BLK;
    const int kc_base = blockIdx.y * KC;

    // loader roles (all 256 threads)
    const int ar = tid >> 2;          // A row within block (0..63)
    const int aq = tid & 3;           // 8-float chunk (0..3)
    const int bn = tid >> 1;          // B column (0..127)
    const int bh = tid & 1;           // 16-element half of k-panel

    // wave / fragment roles (ISA 7.12.2 layouts)
    const int lane = tid & 31;
    const int wv   = tid >> 5;        // 0..7
    const int hf   = lane >> 4;       // 0: lanes 0-15, 1: lanes 16-31
    const int lr   = lane & 15;
    const int mt0  = (wv & 1) * 2;    // first of 2 m-tiles
    const int nt0  = (wv >> 1) * 2;   // first of 2 n-tiles

    v8f acc00 = {}, acc01 = {}, acc10 = {}, acc11 = {};

    const float*  Arow = A   + (size_t)(m_base + ar) * K_DIM + kc_base + aq * 8;
    const __bf16* Bpan = Bws + (size_t)(kc_base >> 5) * (NCOL * 32) + bn * 32 + bh * 16;

    const int adst = ar * 32 + aq * 8;    // bf16 elements into Ash
    const int bdst = bn * 32 + bh * 16;   // bf16 elements into Bsh

    // ---- prologue: panel 0 ----
    copy_b128_to_lds(Bpan,     &Bsh[0][bdst]);
    copy_b128_to_lds(Bpan + 8, &Bsh[0][bdst + 8]);
    {
        float4 p0 = *(const float4*)(Arow);
        float4 p1 = *(const float4*)(Arow + 4);
        *(v8bf*)&Ash[0][adst] = pack8(p0, p1);
    }
    wait_async_lds();
    __syncthreads();

    float4 a0, a1;
    for (int kp = 0; kp < NKP; ++kp) {
        const int  cur  = kp & 1;
        const int  nxt  = cur ^ 1;
        const bool more = (kp + 1 < NKP);

        // issue next panel early (hidden behind WMMAs)
        if (more) {
            const __bf16* Bp = Bpan + (size_t)(kp + 1) * (NCOL * 32);
            copy_b128_to_lds(Bp,     &Bsh[nxt][bdst]);
            copy_b128_to_lds(Bp + 8, &Bsh[nxt][bdst + 8]);
            const float* Ap = Arow + (size_t)(kp + 1) * 32;
            a0 = *(const float4*)(Ap);
            a1 = *(const float4*)(Ap + 4);
        }
        if (kp + 5 < NKP)   // keep the A stream ahead in L2 (global_prefetch_b8)
            __builtin_prefetch(Arow + (size_t)(kp + 5) * 32, 0, 0);

        // A frag: lane<16 holds row lr, K {0..7}+{16..23}; lane>=16 K {8..15}+{24..31}
        union { v16bf v; int4 q[2]; } af0, af1, bg0, bg1;
        const __bf16* Ab = &Ash[cur][0];
        const __bf16* Bb = &Bsh[cur][0];
        const int arow0 = ((mt0    ) * 16 + lr) * 32 + hf * 8;
        const int arow1 = ((mt0 + 1) * 16 + lr) * 32 + hf * 8;
        af0.q[0] = *(const int4*)(Ab + arow0);
        af0.q[1] = *(const int4*)(Ab + arow0 + 16);
        af1.q[0] = *(const int4*)(Ab + arow1);
        af1.q[1] = *(const int4*)(Ab + arow1 + 16);
        // B frag: lane holds column, K {0..7}+{16..23} (lanes>=16 get K {8..15}+{24..31})
        const int bcol0 = ((nt0    ) * 16 + lr) * 32 + hf * 8;
        const int bcol1 = ((nt0 + 1) * 16 + lr) * 32 + hf * 8;
        bg0.q[0] = *(const int4*)(Bb + bcol0);
        bg0.q[1] = *(const int4*)(Bb + bcol0 + 16);
        bg1.q[0] = *(const int4*)(Bb + bcol1);
        bg1.q[1] = *(const int4*)(Bb + bcol1 + 16);

        acc00 = __builtin_amdgcn_wmma_f32_16x16x32_bf16(false, af0.v, false, bg0.v,
                                                        (short)0, acc00, false, false);
        acc01 = __builtin_amdgcn_wmma_f32_16x16x32_bf16(false, af0.v, false, bg1.v,
                                                        (short)0, acc01, false, false);
        acc10 = __builtin_amdgcn_wmma_f32_16x16x32_bf16(false, af1.v, false, bg0.v,
                                                        (short)0, acc10, false, false);
        acc11 = __builtin_amdgcn_wmma_f32_16x16x32_bf16(false, af1.v, false, bg1.v,
                                                        (short)0, acc11, false, false);

        if (more)
            *(v8bf*)&Ash[nxt][adst] = pack8(a0, a1);  // cvt'd A into next buffer
        wait_async_lds();                             // B panel kp+1 landed
        __syncthreads();                              // single barrier per panel
    }

    // Epilogue: C layout (ISA 7.12.2): VGPR j -> row j (lanes 0-15) / j+8 (16-31)
    const int col0 = (nt0    ) * 16 + lr;
    const int col1 = (nt0 + 1) * 16 + lr;
#pragma unroll
    for (int j = 0; j < 8; ++j) {
        const int r0 = m_base + (mt0    ) * 16 + hf * 8 + j;
        const int r1 = m_base + (mt0 + 1) * 16 + hf * 8 + j;
        atomicAdd(&out[r0 * NCOL + col0], acc00[j]);
        atomicAdd(&out[r0 * NCOL + col1], acc01[j]);
        atomicAdd(&out[r1 * NCOL + col0], acc10[j]);
        atomicAdd(&out[r1 * NCOL + col1], acc11[j]);
    }
}

// ---------------------------------------------------------------------------
extern "C" void kernel_launch(void* const* d_in, const int* in_sizes, int n_in,
                              void* d_out, int out_size, void* d_ws, size_t ws_size,
                              hipStream_t stream) {
    const float* x   = (const float*)d_in[0];   // (4096, 32, 4)
    const float* adj = (const float*)d_in[1];   // (4096, 4096, 4)
    const float* w   = (const float*)d_in[2];   // (32, 32, 4)
    float* out  = (float*)d_out;                // (4096, 32, 4) == (4096,128)
    __bf16* Bws = (__bf16*)d_ws;                // needs 4 MB

    // Phase 1: B matrix (16384x128 bf16) + zero output. 2M threads.
    atgco_build_B<<<(K_DIM * NCOL) / 256, 256, 0, stream>>>(x, w, Bws, out);

    // Phase 2: the GEMM. 64 M-blocks x 2 K-chunks = 128 workgroups.
    atgco_gemm_bf16<<<dim3(M_DIM / M_BLK, KSPLIT), 256, 0, stream>>>(adj, Bws, out);
}
